// EncoderLayer_87608742904541
// MI455X (gfx1250) — compile-verified, matrix operand order checked by reading, not run on hardware
//
#include <hip/hip_runtime.h>

// ---------------------------------------------------------------------------
// MI455X / gfx1250 transformer encoder layer (pre-norm, iter=1 path).
// GEMM math via v_wmma_f32_16x16x32_bf16 (wave32). LDS tiles are filled by
// the Tensor Data Mover (tensor_load_to_lds / s_wait_tensorcnt) with a
// double-buffered pipeline: DMA for step s+1 overlaps WMMA on step s.
// ---------------------------------------------------------------------------

typedef __attribute__((ext_vector_type(16))) __bf16 v16bf;
typedef __attribute__((ext_vector_type(8)))  __bf16 v8bf;
typedef __attribute__((ext_vector_type(8)))  float  v8f;
typedef __attribute__((ext_vector_type(4)))  unsigned int u32x4;
typedef __attribute__((ext_vector_type(8)))  int i32x8;
typedef __attribute__((ext_vector_type(4)))  int i32x4;

#define B_SZ   8
#define S_LEN  1024
#define DM     1024
#define DFF    4096
#define NH     16
#define DH     64

#if __has_builtin(__builtin_amdgcn_tensor_load_to_lds) && \
    __has_builtin(__builtin_amdgcn_s_wait_tensorcnt)
#define HAS_TDM 1
#endif

#ifdef HAS_TDM
#if __clang_major__ >= 23
#define TDM_LOAD(g0, g1, g2, g3) \
    __builtin_amdgcn_tensor_load_to_lds((g0), (g1), (g2), (g3), (i32x8){}, 0)
#else
#define TDM_LOAD(g0, g1, g2, g3) \
    __builtin_amdgcn_tensor_load_to_lds((g0), (g1), (g2), (g3), 0)
#endif

__device__ __forceinline__ unsigned lds_off(const void* p) {
    // flat shared pointer -> LDS byte offset (aperture maps addr[31:0] to LDS)
    return (unsigned)(size_t)p;
}

// Issue a 2-D TDM tile load: tile_h rows x tile_w bf16 elements, row stride
// `stride` elements, destination LDS linear row-major (tile_h*tile_w).
// D# layout per CDNA5 ISA ch.8 (group0 128b, group1 256b; groups 2/3 zero).
__device__ __forceinline__ void tdm_load_2d(unsigned ldsaddr, const void* gptr,
                                            int tile_w, int tile_h, int stride) {
    unsigned long long ga = (unsigned long long)(size_t)gptr;
    u32x4 g0;
    g0[0] = 1u;                                  // count=1, user descriptor
    g0[1] = ldsaddr;                             // lds_addr
    g0[2] = (unsigned)(ga & 0xFFFFFFFFu);        // global_addr[31:0]
    g0[3] = (unsigned)((ga >> 32) & 0x01FFFFFFu) // global_addr[56:32]
            | (2u << 30);                        // type=2 ("image")
    unsigned tw = (unsigned)tile_w, th = (unsigned)tile_h, st = (unsigned)stride;
    i32x8 g1;
    g1[0] = (int)(1u << 16);                     // data_size=1 (2-byte elements)
    g1[1] = (int)((tw & 0xFFFFu) << 16);         // tensor_dim0[15:0]  @ bits 63:48
    g1[2] = (int)(((tw >> 16) & 0xFFFFu) | ((th & 0xFFFFu) << 16)); // d0 hi | d1 lo
    g1[3] = (int)(((th >> 16) & 0xFFFFu) | ((tw & 0xFFFFu) << 16)); // d1 hi | tile_dim0
    g1[4] = (int)(th & 0xFFFFu);                 // tile_dim1 (tile_dim2 = 0)
    g1[5] = (int)st;                             // tensor_dim0_stride[31:0]
    g1[6] = 0;                                   // stride0[47:32] | stride1 lo
    g1[7] = 0;
    i32x4 z = {0, 0, 0, 0};
    TDM_LOAD(g0, g1, z, z);
}
#endif  // HAS_TDM

// float -> bf16, round-to-nearest-even
__device__ __forceinline__ __bf16 f2bf(float f) {
    unsigned u = __builtin_bit_cast(unsigned, f);
    unsigned r = (u + 0x7FFFu + ((u >> 16) & 1u)) >> 16;
    unsigned short h = (unsigned short)r;
    return __builtin_bit_cast(__bf16, h);
}

__device__ __forceinline__ float tanh_fast(float a) {
#if __has_builtin(__builtin_amdgcn_tanhf)
    return __builtin_amdgcn_tanhf(a);            // native v_tanh_f32 on gfx1250
#elif __has_builtin(__builtin_amdgcn_tanh_f32)
    return __builtin_amdgcn_tanh_f32(a);
#else
    float e = __expf(2.0f * a);                  // branch-free fallback
    return (e - 1.0f) / (e + 1.0f);
#endif
}

__device__ __forceinline__ float gelu_f(float x) {
    const float c = 0.7978845608028654f;  // sqrt(2/pi)
    float t = tanh_fast(c * (x + 0.044715f * x * x * x));
    return 0.5f * x * (1.0f + t);
}

__device__ __forceinline__ v8f wmma_bf16(v16bf a, v16bf b, v8f c) {
    return __builtin_amdgcn_wmma_f32_16x16x32_bf16(
        false, a, false, b, (short)0, c, false, false);
}

// A-fragment (or B-fragment whose columns are contiguous rows, e.g. K^T):
// per-lane K layout: base = (lane>=16)?8:0; j<8 -> K=base+j; j>=8 -> K=16+base+j-8
__device__ __forceinline__ v16bf load_row_frag(const __bf16* t, int row, int ld,
                                               int kbase, int lane) {
    int base = kbase + ((lane & 16) ? 8 : 0);
    const __bf16* p = t + row * ld + base;
    v8bf lo = *(const v8bf*)(p);
    v8bf hi = *(const v8bf*)(p + 16);
    return __builtin_shufflevector(lo, hi, 0,1,2,3,4,5,6,7,8,9,10,11,12,13,14,15);
}

// B-fragment for a row-major [K x N] LDS tile: gather column `col`.
__device__ __forceinline__ v16bf load_col_frag(const __bf16* t, int col, int ld,
                                               int kbase, int lane) {
    int base = kbase + ((lane & 16) ? 8 : 0);
    v16bf r;
#pragma unroll
    for (int j = 0; j < 16; ++j) {
        int k = base + j + ((j & 8) ? 8 : 0);
        r[j] = t[k * ld + col];
    }
    return r;
}

// ---------------------------------------------------------------------------
__global__ void cast_bf16_kernel(const float* __restrict__ in,
                                 __bf16* __restrict__ out, int n) {
    int i = blockIdx.x * 256 + threadIdx.x;
    if (i < n) out[i] = f2bf(in[i]);
}

// ---------------------------------------------------------------------------
// LayerNorm over D=1024, optional fused residual add.
// ---------------------------------------------------------------------------
__global__ __launch_bounds__(256)
void ln_kernel(const float* __restrict__ in, const float* __restrict__ res,
               const float* __restrict__ g, const float* __restrict__ bta,
               float* __restrict__ sum_out, __bf16* __restrict__ nout) {
    const int D = DM;
    const int row = blockIdx.x;
    const int t = threadIdx.x;
    __shared__ float red[256];

    const size_t base = (size_t)row * D;
    float v[4];
    float s = 0.f;
#pragma unroll
    for (int i = 0; i < 4; ++i) {
        int c = t + i * 256;
        float x = in[base + c];
        if (res) x += res[base + c];
        v[i] = x;
        s += x;
    }
    red[t] = s;
    __syncthreads();
    for (int o = 128; o > 0; o >>= 1) {
        if (t < o) red[t] += red[t + o];
        __syncthreads();
    }
    float mu = red[0] * (1.0f / D);
    __syncthreads();
    float sq = 0.f;
#pragma unroll
    for (int i = 0; i < 4; ++i) { float d = v[i] - mu; sq += d * d; }
    red[t] = sq;
    __syncthreads();
    for (int o = 128; o > 0; o >>= 1) {
        if (t < o) red[t] += red[t + o];
        __syncthreads();
    }
    float rstd = rsqrtf(red[0] * (1.0f / D) + 1e-6f);
#pragma unroll
    for (int i = 0; i < 4; ++i) {
        int c = t + i * 256;
        if (sum_out) sum_out[base + c] = v[i];
        nout[base + c] = f2bf((v[i] - mu) * rstd * g[c] + bta[c]);
    }
}

// One 128x128x32 block-tile step: fragment loads + 8 WMMAs for this wave.
__device__ __forceinline__ void gemm_step(const __bf16* As, const __bf16* Bs,
                                          v8f (&acc)[2][4],
                                          int wr, int wc, int lh, int lane) {
    v16bf af[2], bfr[4];
    af[0] = load_row_frag(As, wr + lh, 32, 0, lane);
    af[1] = load_row_frag(As, wr + 16 + lh, 32, 0, lane);
#pragma unroll
    for (int n = 0; n < 4; ++n)
        bfr[n] = load_col_frag(Bs, wc + n * 16 + lh, 128, 0, lane);
#pragma unroll
    for (int m = 0; m < 2; ++m)
#pragma unroll
        for (int n = 0; n < 4; ++n)
            acc[m][n] = wmma_bf16(af[m], bfr[n], acc[m][n]);
}

// ---------------------------------------------------------------------------
// Generic bf16 WMMA GEMM: C = epi((A@B + bias) * scale) [+ res]
// Block tile 128x128, K-step 32; 8 waves x (2x4) WMMA tiles.
// TDM path: double-buffered — wave 0 issues the DMA for step s+1 before the
// block computes step s, and drains TENSORcnt only after the math.
// ---------------------------------------------------------------------------
__global__ __launch_bounds__(256)
void gemm_bf16_kernel(const __bf16* __restrict__ A, const __bf16* __restrict__ B,
                      const float* __restrict__ bias, const float* __restrict__ res,
                      float* __restrict__ outF, __bf16* __restrict__ outB,
                      int M, int N, int K, float scale, int epi) {
    __shared__ __bf16 As[2][128 * 32];   // 2 x 8 KB
    __shared__ __bf16 Bs[2][32 * 128];   // 2 x 8 KB

    const int tid = threadIdx.x;
    const int w = tid >> 5, lane = tid & 31;
    const int lh = lane & 15, hb = lane >> 4;
    const int bm = blockIdx.y * 128, bn = blockIdx.x * 128;
    const int wr = (w >> 1) * 32, wc = (w & 1) * 64;

    v8f acc[2][4];
#pragma unroll
    for (int m = 0; m < 2; ++m)
#pragma unroll
        for (int n = 0; n < 4; ++n) acc[m][n] = (v8f){};

    const int nsteps = K >> 5;

#ifdef HAS_TDM
    if (w == 0) {   // prologue: fill buffer 0
        tdm_load_2d(lds_off(As[0]), A + (size_t)bm * K, 32, 128, K);
        tdm_load_2d(lds_off(Bs[0]), B + bn, 128, 32, N);
        __builtin_amdgcn_s_wait_tensorcnt(0);
    }
    __syncthreads();
    for (int s = 0; s < nsteps; ++s) {
        const int cur = s & 1;
        const bool more = (s + 1) < nsteps;
        if (w == 0 && more) {   // overlap: DMA next tiles into the other buffer
            int kt = (s + 1) << 5;
            tdm_load_2d(lds_off(As[cur ^ 1]), A + (size_t)bm * K + kt, 32, 128, K);
            tdm_load_2d(lds_off(Bs[cur ^ 1]), B + (size_t)kt * N + bn, 128, 32, N);
        }
        gemm_step(As[cur], Bs[cur], acc, wr, wc, lh, lane);
        if (w == 0 && more) __builtin_amdgcn_s_wait_tensorcnt(0);
        __syncthreads();
    }
#else
    for (int s = 0; s < nsteps; ++s) {
        const int kt = s << 5;
        {   // stage A tile 128x32
            int r = tid >> 1, sg = tid & 1;
            *(v8bf*)&As[0][r * 32 + sg * 16] =
                *(const v8bf*)&A[(size_t)(bm + r) * K + kt + sg * 16];
            if (kt + 32 < K)
                __builtin_prefetch(&A[(size_t)(bm + r) * K + kt + 32 + sg * 16], 0, 3);
        }
        {   // stage B tile 32x128
            int r = tid >> 3, sg = tid & 7;
            *(v8bf*)&Bs[0][r * 128 + sg * 16] =
                *(const v8bf*)&B[(size_t)(kt + r) * N + bn + sg * 16];
            if (kt + 32 < K)
                __builtin_prefetch(&B[(size_t)(kt + 32 + r) * N + bn + sg * 16], 0, 3);
        }
        __syncthreads();
        gemm_step(As[0], Bs[0], acc, wr, wc, lh, lane);
        __syncthreads();
    }
#endif

#pragma unroll
    for (int m = 0; m < 2; ++m)
#pragma unroll
        for (int n = 0; n < 4; ++n)
#pragma unroll
            for (int r = 0; r < 8; ++r) {
                int row = bm + wr + m * 16 + hb * 8 + r;
                int col = bn + wc + n * 16 + lh;
                float v = (acc[m][n][r] + bias[col]) * scale;
                if (epi == 1) v = gelu_f(v);
                if (res) v += res[(size_t)row * N + col];
                size_t o = (size_t)row * N + col;
                if (outF) outF[o] = v;
                if (outB) outB[o] = f2bf(v);
            }
}

// ---------------------------------------------------------------------------
// Fused flash-style attention for one (b,h): online softmax over the
// *multiplicative* tril mask (masked logits are exactly 0.0 and still feed
// the softmax — faithful to the reference). Each wave owns 16 q rows, so the
// softmax reductions are width-16 shuffles. Output scattered through the
// context-reshape bug permutation.
// ---------------------------------------------------------------------------
__global__ __launch_bounds__(256)
void flash_attn_kernel(const __bf16* __restrict__ Q, const __bf16* __restrict__ K,
                       const __bf16* __restrict__ V, float* __restrict__ ctx) {
    __shared__ __bf16 Kt[128 * 64];      // 16 KB
    __shared__ __bf16 Vt[128 * 64];      // 16 KB
    __shared__ __bf16 Pst[8][16 * 32];   // 8 KB (per-wave P restage)

    const int tid = threadIdx.x;
    const int w = tid >> 5, lane = tid & 31;
    const int lh = lane & 15, hb = lane >> 4;
    const int bh = blockIdx.y;
    const int b = bh >> 4, h = bh & 15;
    const int q0 = blockIdx.x * 128;

    const size_t hbase = (size_t)b * S_LEN * DM + (size_t)h * DH;
    const __bf16* Qg = Q + hbase;
    const __bf16* Kg = K + hbase;
    const __bf16* Vg = V + hbase;

    // Stage Q tile (borrow Kt), pull per-wave Q fragments (d = 2 k-steps)
#ifdef HAS_TDM
    if (w == 0) {
        tdm_load_2d(lds_off(Kt), Qg + (size_t)q0 * DM, 64, 128, DM);
        __builtin_amdgcn_s_wait_tensorcnt(0);
    }
#else
    for (int c = tid; c < 512; c += 256) {
        int r = c >> 2, sg = c & 3;
        *(v8bf*)&Kt[r * 64 + sg * 16] =
            *(const v8bf*)&Qg[(size_t)(q0 + r) * DM + sg * 16];
    }
#endif
    __syncthreads();
    v16bf qf[2];
    qf[0] = load_row_frag(Kt, w * 16 + lh, 64, 0, lane);
    qf[1] = load_row_frag(Kt, w * 16 + lh, 64, 32, lane);
    __syncthreads();

    float m_run[8], l_run[8];
    v8f O[4];
#pragma unroll
    for (int r = 0; r < 8; ++r) { m_run[r] = -3.0e38f; l_run[r] = 0.f; }
#pragma unroll
    for (int n = 0; n < 4; ++n) O[n] = (v8f){};

    for (int j = 0; j < 8; ++j) {   // all k blocks (masked logits are 0, kept)
        const int k0 = j * 128;
#ifdef HAS_TDM
        if (w == 0) {
            tdm_load_2d(lds_off(Kt), Kg + (size_t)k0 * DM, 64, 128, DM);
            tdm_load_2d(lds_off(Vt), Vg + (size_t)k0 * DM, 64, 128, DM);
            __builtin_amdgcn_s_wait_tensorcnt(0);
        }
#else
        for (int c = tid; c < 512; c += 256) {
            int r = c >> 2, sg = c & 3;
            *(v8bf*)&Kt[r * 64 + sg * 16] =
                *(const v8bf*)&Kg[(size_t)(k0 + r) * DM + sg * 16];
            *(v8bf*)&Vt[r * 64 + sg * 16] =
                *(const v8bf*)&Vg[(size_t)(k0 + r) * DM + sg * 16];
        }
#endif
        __syncthreads();

        // S = Q * K^T  (columns of K^T are K rows -> row loader)
        v8f s[8];
#pragma unroll
        for (int c = 0; c < 8; ++c) {
            v16bf kf0 = load_row_frag(Kt, c * 16 + lh, 64, 0, lane);
            v16bf kf1 = load_row_frag(Kt, c * 16 + lh, 64, 32, lane);
            v8f a = (v8f){};
            a = wmma_bf16(qf[0], kf0, a);
            a = wmma_bf16(qf[1], kf1, a);
            s[c] = a;
        }

        // multiplicative tril mask + online softmax
#pragma unroll
        for (int r = 0; r < 8; ++r) {
            int qg = q0 + w * 16 + hb * 8 + r;
            float rm = -3.0e38f;
#pragma unroll
            for (int c = 0; c < 8; ++c) {
                int kg = k0 + c * 16 + lh;
                float vv = (kg <= qg) ? s[c][r] : 0.0f;   // score * mask
                s[c][r] = vv;
                rm = fmaxf(rm, vv);
            }
            rm = fmaxf(rm, __shfl_xor(rm, 1, 16));
            rm = fmaxf(rm, __shfl_xor(rm, 2, 16));
            rm = fmaxf(rm, __shfl_xor(rm, 4, 16));
            rm = fmaxf(rm, __shfl_xor(rm, 8, 16));
            float mn = fmaxf(m_run[r], rm);
            float sc = __expf(m_run[r] - mn);
            m_run[r] = mn;
            float rs = 0.f;
#pragma unroll
            for (int c = 0; c < 8; ++c) {
                float p = __expf(s[c][r] - mn);
                s[c][r] = p;
                rs += p;
            }
            rs += __shfl_xor(rs, 1, 16);
            rs += __shfl_xor(rs, 2, 16);
            rs += __shfl_xor(rs, 4, 16);
            rs += __shfl_xor(rs, 8, 16);
            l_run[r] = l_run[r] * sc + rs;
#pragma unroll
            for (int n = 0; n < 4; ++n) O[n][r] *= sc;
        }

        // O += P * V : restage P (C-layout) -> A-fragment layout via LDS
#pragma unroll
        for (int ks = 0; ks < 4; ++ks) {
            __bf16* pp = &Pst[w][0];
#pragma unroll
            for (int cc = 0; cc < 2; ++cc) {
                int c = ks * 2 + cc;
#pragma unroll
                for (int r = 0; r < 8; ++r)
                    pp[(hb * 8 + r) * 32 + cc * 16 + lh] = f2bf(s[c][r]);
            }
            v16bf pf = load_row_frag(pp, lh, 32, 0, lane);
#pragma unroll
            for (int n = 0; n < 4; ++n) {
                v16bf vf = load_col_frag(Vt, n * 16 + lh, 64, ks * 32, lane);
                O[n] = wmma_bf16(pf, vf, O[n]);
            }
        }
        __syncthreads();
    }

    // normalize + scatter with the context-reshape bug:
    //   out[b, h*64 + s/16, (s%16)*64 + d] = ctx[b,h,s,d]
#pragma unroll
    for (int r = 0; r < 8; ++r) {
        float inv = 1.0f / l_run[r];
        int sidx = q0 + w * 16 + hb * 8 + r;
        int sp = h * 64 + (sidx >> 4);
        int dp0 = (sidx & 15) << 6;
#pragma unroll
        for (int n = 0; n < 4; ++n) {
            int d = n * 16 + lh;
            ctx[(size_t)b * S_LEN * DM + (size_t)sp * DM + dp0 + d] = O[n][r] * inv;
        }
    }
}

// ---------------------------------------------------------------------------
// Host-side orchestration (all launches on `stream`; graph-capture safe)
// ---------------------------------------------------------------------------
extern "C" void kernel_launch(void* const* d_in, const int* in_sizes, int n_in,
                              void* d_out, int out_size, void* d_ws, size_t ws_size,
                              hipStream_t stream) {
    (void)in_sizes; (void)n_in; (void)out_size; (void)ws_size;
    const float* x        = (const float*)d_in[1];
    const float* Wq       = (const float*)d_in[3];
    const float* bq       = (const float*)d_in[4];
    const float* Wk       = (const float*)d_in[5];
    const float* bk       = (const float*)d_in[6];
    const float* Wv       = (const float*)d_in[7];
    const float* bv       = (const float*)d_in[8];
    const float* W1       = (const float*)d_in[9];
    const float* b1       = (const float*)d_in[10];
    const float* W2       = (const float*)d_in[11];
    const float* b2       = (const float*)d_in[12];
    const float* ln_a_g   = (const float*)d_in[13];
    const float* ln_a_b   = (const float*)d_in[14];
    const float* ln_f_g   = (const float*)d_in[15];
    const float* ln_f_b   = (const float*)d_in[16];

    const int M = B_SZ * S_LEN;             // 8192
    const size_t actE = (size_t)M * DM;

    char* ws = (char*)d_ws;
    size_t off = 0;
    auto alloc = [&](size_t bytes) -> void* {
        void* p = ws + off;
        off += (bytes + 255) & ~(size_t)255;
        return p;
    };

    __bf16* nrm  = (__bf16*)alloc(actE * 2);
    __bf16* Qb   = (__bf16*)alloc(actE * 2);
    __bf16* Kb   = (__bf16*)alloc(actE * 2);
    __bf16* Vb   = (__bf16*)alloc(actE * 2);
    __bf16* WqB  = (__bf16*)alloc((size_t)DM * DM * 2);
    __bf16* WkB  = (__bf16*)alloc((size_t)DM * DM * 2);
    __bf16* WvB  = (__bf16*)alloc((size_t)DM * DM * 2);
    __bf16* W1B  = (__bf16*)alloc((size_t)DM * DFF * 2);
    __bf16* W2B  = (__bf16*)alloc((size_t)DFF * DM * 2);
    float*  ctx  = (float*)alloc(actE * 4);
    float*  outb = (float*)alloc(actE * 4);
    __bf16* hnrm = (__bf16*)alloc(actE * 2);
    __bf16* h1   = (__bf16*)alloc((size_t)M * DFF * 2);

    {   // weights -> bf16
        int n = DM * DM;
        cast_bf16_kernel<<<n / 256, 256, 0, stream>>>(Wq, WqB, n);
        cast_bf16_kernel<<<n / 256, 256, 0, stream>>>(Wk, WkB, n);
        cast_bf16_kernel<<<n / 256, 256, 0, stream>>>(Wv, WvB, n);
        int nf = DM * DFF;
        cast_bf16_kernel<<<nf / 256, 256, 0, stream>>>(W1, W1B, nf);
        cast_bf16_kernel<<<nf / 256, 256, 0, stream>>>(W2, W2B, nf);
    }

    // pre-norm (iter=1 path)
    ln_kernel<<<M, 256, 0, stream>>>(x, nullptr, ln_a_g, ln_a_b, nullptr, nrm);

    // Q,K,V projections (Q fused with 1/sqrt(Dh) = 0.125)
    dim3 gqkv(DM / 128, M / 128);
    gemm_bf16_kernel<<<gqkv, 256, 0, stream>>>(nrm, WqB, bq, nullptr, nullptr, Qb,
                                               M, DM, DM, 0.125f, 0);
    gemm_bf16_kernel<<<gqkv, 256, 0, stream>>>(nrm, WkB, bk, nullptr, nullptr, Kb,
                                               M, DM, DM, 1.0f, 0);
    gemm_bf16_kernel<<<gqkv, 256, 0, stream>>>(nrm, WvB, bv, nullptr, nullptr, Vb,
                                               M, DM, DM, 1.0f, 0);

    // fused attention -> ctx (already in reshape-bug layout)
    dim3 gat(S_LEN / 128, B_SZ * NH);
    flash_attn_kernel<<<gat, 256, 0, stream>>>(Qb, Kb, Vb, ctx);

    // residual add + FFN pre-norm
    ln_kernel<<<M, 256, 0, stream>>>(ctx, x, ln_f_g, ln_f_b, outb, hnrm);

    // FFN1 (GELU) and FFN2 (+bias +residual) -> final f32 output
    dim3 gf1(DFF / 128, M / 128);
    gemm_bf16_kernel<<<gf1, 256, 0, stream>>>(hnrm, W1B, b1, nullptr, nullptr, h1,
                                              M, DFF, DM, 1.0f, 1);
    dim3 gf2(DM / 128, M / 128);
    gemm_bf16_kernel<<<gf2, 256, 0, stream>>>(h1, W2B, b2, outb, (float*)d_out,
                                              nullptr, M, DM, DFF, 1.0f, 0);
}
